// SBERT_66992899883323
// MI455X (gfx1250) — matrix operand order repaired and checked by reading, not compiled, not run on hardware
//
#include <hip/hip_runtime.h>
#include <hip/hip_bf16.h>
#include <stdint.h>
#include <math.h>

typedef __bf16 bf16;
typedef __attribute__((ext_vector_type(16))) __bf16 v16bf;
typedef __attribute__((ext_vector_type(8)))  float  v8f;

namespace {
constexpr int kL   = 6;
constexpr int kD   = 1024;
constexpr int kH   = 16;
constexpr int kFin = 128;
constexpr int kB   = 8;
constexpr int kS   = 512;
constexpr int kDff = 4096;
constexpr int kDk  = 64;
constexpr int kD2  = 512;
constexpr int kRows = kB * kS;                 // 4096 token rows
constexpr long long kSS = (long long)kS * kS;  // 262144
}

// gfx1250 async global->LDS copy (ASYNCcnt-tracked); IOFFSET applies to both
// the LDS and global addresses per the CDNA5 ISA, so one base covers 32B.
#define ASYNC_LDS_B128(ldsb, gptr)                                            \
  asm volatile("global_load_async_to_lds_b128 %0, %1, off"                    \
               :: "v"(ldsb), "v"((unsigned long long)(uintptr_t)(gptr))       \
               : "memory")
#define ASYNC_LDS_B128_O16(ldsb, gptr)                                        \
  asm volatile("global_load_async_to_lds_b128 %0, %1, off offset:16"          \
               :: "v"(ldsb), "v"((unsigned long long)(uintptr_t)(gptr))       \
               : "memory")

__device__ __forceinline__ uint32_t lds_addr_of(const void* p) {
  return (uint32_t)(uintptr_t)((__attribute__((address_space(3))) const void*)p);
}

__device__ __forceinline__ void wait_async_lds() {
#if __has_builtin(__builtin_amdgcn_s_wait_asynccnt)
  __builtin_amdgcn_s_wait_asynccnt(0);
#else
  asm volatile("s_wait_asynccnt 0x0" ::: "memory");
#endif
}

// ---------------------------------------------------------------------------
// fp32 -> bf16 staging
// ---------------------------------------------------------------------------
__global__ __launch_bounds__(256) void f32_to_bf16_k(const float* __restrict__ in,
                                                     bf16* __restrict__ out, int n) {
  int i = blockIdx.x * 256 + threadIdx.x;
  if (i < n) out[i] = (bf16)in[i];
}

__global__ __launch_bounds__(256) void copy_f32_k(const float* __restrict__ in,
                                                  float* __restrict__ out, int n) {
  int i = blockIdx.x * 256 + threadIdx.x;
  if (i < n) out[i] = in[i];
}

// ---------------------------------------------------------------------------
// Embedding: h[:, :512] = x @ w_in + b_in ; h[:, 512:] = pe[doy]
// ---------------------------------------------------------------------------
__global__ __launch_bounds__(256) void embed_k(const float* __restrict__ x,
                                               const int* __restrict__ doy,
                                               const float* __restrict__ w_in,
                                               const float* __restrict__ b_in,
                                               const float* __restrict__ pe,
                                               float* __restrict__ h) {
  __shared__ float xr[kFin];
  const int row = blockIdx.x;
  const int tid = threadIdx.x;
  if (tid < kFin) xr[tid] = x[(size_t)row * kFin + tid];
  __syncthreads();
  for (int j = tid; j < kD2; j += 256) {
    float acc = b_in[j];
    #pragma unroll 8
    for (int k = 0; k < kFin; ++k) acc += xr[k] * w_in[(size_t)k * kD2 + j];
    h[(size_t)row * kD + j] = acc;
  }
  const int d = doy[row];
  for (int j = tid; j < kD2; j += 256)
    h[(size_t)row * kD + kD2 + j] = pe[(size_t)d * kD2 + j];
}

// ---------------------------------------------------------------------------
// LayerNorm over D=1024, writes bf16 GEMM A-operand. One block per row.
// ---------------------------------------------------------------------------
__global__ __launch_bounds__(256) void layernorm_k(const float* __restrict__ h,
                                                   const float* __restrict__ g,
                                                   const float* __restrict__ b,
                                                   bf16* __restrict__ xn) {
  __shared__ float rs[256], rq[256];
  const int row = blockIdx.x, tid = threadIdx.x;
  float v[4], s = 0.f, s2 = 0.f;
  #pragma unroll
  for (int j = 0; j < 4; ++j) {
    v[j] = h[(size_t)row * kD + tid + j * 256];
    s += v[j]; s2 += v[j] * v[j];
  }
  rs[tid] = s; rq[tid] = s2;
  __syncthreads();
  for (int st = 128; st > 0; st >>= 1) {
    if (tid < st) { rs[tid] += rs[tid + st]; rq[tid] += rq[tid + st]; }
    __syncthreads();
  }
  const float mean = rs[0] * (1.f / kD);
  const float var  = rq[0] * (1.f / kD) - mean * mean;
  const float rstd = rsqrtf(var + 1e-5f);
  #pragma unroll
  for (int j = 0; j < 4; ++j) {
    const int c = tid + j * 256;
    xn[(size_t)row * kD + c] = (bf16)((v[j] - mean) * rstd * g[c] + b[c]);
  }
}

// ---------------------------------------------------------------------------
// Faithful quirk: softmax over the HEAD axis. One thread per (b, q, k).
// ---------------------------------------------------------------------------
__global__ __launch_bounds__(256) void softmax_head_k(const float* __restrict__ scores,
                                                      const int* __restrict__ mask,
                                                      bf16* __restrict__ p) {
  const long long idx = (long long)blockIdx.x * 256 + threadIdx.x;
  if (idx >= (long long)kB * kS * kS) return;
  const int kc = (int)(idx % kS);
  const int q  = (int)((idx / kS) % kS);
  const int b  = (int)(idx / ((long long)kS * kS));
  const long long base = ((long long)b * kH) * kSS + (long long)q * kS + kc;
  if (mask[b * kS + kc] > 0) {
    float vals[kH], mx = -3.4e38f;
    #pragma unroll
    for (int hh = 0; hh < kH; ++hh) {
      vals[hh] = scores[base + (long long)hh * kSS];
      mx = fmaxf(mx, vals[hh]);
    }
    float sum = 0.f;
    #pragma unroll
    for (int hh = 0; hh < kH; ++hh) { vals[hh] = __expf(vals[hh] - mx); sum += vals[hh]; }
    const float inv = 1.f / sum;
    #pragma unroll
    for (int hh = 0; hh < kH; ++hh) p[base + (long long)hh * kSS] = (bf16)(vals[hh] * inv);
  } else {
    #pragma unroll
    for (int hh = 0; hh < kH; ++hh) p[base + (long long)hh * kSS] = (bf16)0.0625f;
  }
}

// ---------------------------------------------------------------------------
// WMMA fragment loaders from LDS (tiles stored k-contiguous, 32 bf16/row).
// A 16x32 bf16: lanes 0-15 M rows; K-pairs {0..3}+4h, {8..11}+4h.
// B 32x16 bf16: lanes hold N; K-pairs {0..3}+4h, {8..11}+4h. Both vectorize to ds_load_b128.
// ---------------------------------------------------------------------------
__device__ __forceinline__ v16bf load_frag_a(const bf16* As, int mrow, int half) {
  const uint32_t* p = (const uint32_t*)(As + mrow * 32);
  union { v16bf v; uint32_t u[8]; } f;
  #pragma unroll
  for (int i = 0; i < 8; ++i) {
    const int kp = (i < 4) ? (i + 4 * half) : (8 + (i - 4) + 4 * half);
    f.u[i] = p[kp];
  }
  return f.v;
}

__device__ __forceinline__ v16bf load_frag_b(const bf16* Bs, int ncol, int half) {
  const uint32_t* p = (const uint32_t*)(Bs + ncol * 32);
  union { v16bf v; uint32_t u[8]; } f;
  #pragma unroll
  for (int i = 0; i < 8; ++i) f.u[i] = p[(i < 4) ? (i + 4 * half) : (8 + (i - 4) + 4 * half)];
  return f.v;
}

// ---------------------------------------------------------------------------
// Generic batched bf16 WMMA GEMM:  C = act(alpha * A@B + bias + resid)
//  block = 256 threads (8 wave32s). Block tile 128x128, wave tile 32x64
//  (2x4 v_wmma_f32_16x16x32_bf16), BK = 32. blockIdx.z -> (batch, head).
//  A tile and transposed-B tile staged with GLOBAL_LOAD_ASYNC_TO_LDS_B128;
//  normal-B needs a transpose scatter, so it stages through VGPRs.
//  Requires M % 128 == 0 (true for all uses: 4096 and 512).
// ---------------------------------------------------------------------------
__global__ __launch_bounds__(256)
void gemm_bf16_wmma(const bf16* __restrict__ A, int lda, long long sAb, long long sAh,
                    const bf16* __restrict__ B, int ldb, long long sBb, long long sBh,
                    int btrans,
                    const float* __restrict__ bias,
                    const float* __restrict__ resid,
                    float* __restrict__ Cf, bf16* __restrict__ Cb,
                    int ldc, long long sCb, long long sCh,
                    int M, int N, int K, int Hdim, float alpha, int act) {
  __shared__ alignas(16) bf16 As[128 * 32];   // [m][k]
  __shared__ alignas(16) bf16 Bs[128 * 32];   // [n][k]

  const int tid  = threadIdx.x;
  const int lane = tid & 31;
  const int wave = tid >> 5;
  const int wm   = wave >> 1;                 // 0..3 -> 32-row strip
  const int wn   = wave & 1;                  // 0..1 -> 64-col strip
  const int tileM = blockIdx.y * 128;
  const int tileN = blockIdx.x * 128;
  const int half  = lane >> 4;
  const int lm    = lane & 15;

  const int zb = blockIdx.z / Hdim;
  const int zh = blockIdx.z % Hdim;
  A += (long long)zb * sAb + (long long)zh * sAh;
  B += (long long)zb * sBb + (long long)zh * sBh;
  const long long coff = (long long)zb * sCb + (long long)zh * sCh;

  const uint32_t asBase = lds_addr_of(As);
  const uint32_t bsBase = lds_addr_of(Bs);

  v8f acc[2][4];
  #pragma unroll
  for (int mt = 0; mt < 2; ++mt)
    #pragma unroll
    for (int nt = 0; nt < 4; ++nt)
      acc[mt][nt] = v8f{0,0,0,0,0,0,0,0};

  // per-thread staging coordinates
  const int arow = tid >> 1, ahf = tid & 1;   // A / trans-B: 128 rows x 2 halves

  for (int k0 = 0; k0 < K; k0 += 32) {
    __syncthreads();   // previous compute's LDS reads are complete (dscnt waited)

    // ---- A tile 128x32: async global->LDS, k-contiguous (M%128==0, no guard)
    {
      const uint32_t lA = asBase + (uint32_t)(arow * 32 + ahf * 16) * 2u;
      const bf16* gA = A + (size_t)(tileM + arow) * lda + k0 + ahf * 16;
      ASYNC_LDS_B128(lA, gA);
      ASYNC_LDS_B128_O16(lA, gA);
    }
    // ---- B tile into [n][k] layout ----
    if (btrans) {
      // B element (k,n) = B[n*ldb + k]: rows are k-contiguous -> async friendly
      const uint32_t lB = bsBase + (uint32_t)(arow * 32 + ahf * 16) * 2u;
      const bf16* gB = B + (size_t)(tileN + arow) * ldb + k0 + ahf * 16;
      ASYNC_LDS_B128(lB, gB);
      ASYNC_LDS_B128_O16(lB, gB);
    } else {
      // row-major B: transpose scatter through VGPRs
      const int kk = tid >> 3, seg = tid & 7;
      union { uint4 q[2]; bf16 e[16]; } tv;
      const bf16* src = B + (size_t)(k0 + kk) * ldb + tileN + seg * 16;
      if (tileN + seg * 16 + 16 <= N) {
        tv.q[0] = *(const uint4*)(src);
        tv.q[1] = *(const uint4*)(src + 8);
      } else {
        #pragma unroll
        for (int j = 0; j < 16; ++j)
          tv.e[j] = (tileN + seg * 16 + j < N) ? src[j] : (bf16)0.0f;
      }
      #pragma unroll
      for (int j = 0; j < 16; ++j) Bs[(seg * 16 + j) * 32 + kk] = tv.e[j];
    }
    wait_async_lds();
    __syncthreads();

    // ---- 2x4 WMMA tiles per wave ----
    const v16bf a0 = load_frag_a(As, wm * 32 + lm,      half);
    const v16bf a1 = load_frag_a(As, wm * 32 + 16 + lm, half);
    v16bf bfrag[4];
    #pragma unroll
    for (int nt = 0; nt < 4; ++nt)
      bfrag[nt] = load_frag_b(Bs, wn * 64 + nt * 16 + lm, half);
    #pragma unroll
    for (int nt = 0; nt < 4; ++nt) {
      acc[0][nt] = __builtin_amdgcn_wmma_f32_16x16x32_bf16(false, a0, false, bfrag[nt],
                                                           (short)0, acc[0][nt], false, false);
      acc[1][nt] = __builtin_amdgcn_wmma_f32_16x16x32_bf16(false, a1, false, bfrag[nt],
                                                           (short)0, acc[1][nt], false, false);
    }
  }

  // ---- epilogue: C layout VGPR v -> M = v + 8*half, N = lane&15 ----
  auto store_tile = [&](v8f a, int mbase, int nbase) {
    #pragma unroll
    for (int v = 0; v < 8; ++v) {
      const int gm = mbase + v + 8 * half;
      const int gn = nbase + lm;
      if (gm < M && gn < N) {
        float val = a[v] * alpha;
        if (bias)  val += bias[gn];
        if (resid) val += resid[coff + (size_t)gm * ldc + gn];
        if (act == 1) {
          const float xx = val;
          val = 0.5f * xx * (1.0f + tanhf(0.7978845608028654f * (xx + 0.044715f * xx * xx * xx)));
        }
        if (Cf) Cf[coff + (size_t)gm * ldc + gn] = val;
        if (Cb) Cb[coff + (size_t)gm * ldc + gn] = (bf16)val;
      }
    }
  };
  #pragma unroll
  for (int mt = 0; mt < 2; ++mt)
    #pragma unroll
    for (int nt = 0; nt < 4; ++nt)
      store_tile(acc[mt][nt], tileM + wm * 32 + mt * 16, tileN + wn * 64 + nt * 16);
}

// ---------------------------------------------------------------------------
// host driver
// ---------------------------------------------------------------------------
extern "C" void kernel_launch(void* const* d_in, const int* in_sizes, int n_in,
                              void* d_out, int out_size, void* d_ws, size_t ws_size,
                              hipStream_t stream) {
  (void)in_sizes; (void)n_in; (void)out_size; (void)ws_size;
  const float* x     = (const float*)d_in[0];
  const int*   doy   = (const int*)  d_in[1];
  const int*   mask  = (const int*)  d_in[2];
  const float* w_in  = (const float*)d_in[3];
  const float* b_in  = (const float*)d_in[4];
  const float* pe    = (const float*)d_in[5];
  const float* qkv_w = (const float*)d_in[6];
  const float* qkv_b = (const float*)d_in[7];
  const float* out_w = (const float*)d_in[8];
  const float* out_b = (const float*)d_in[9];
  const float* ln1g  = (const float*)d_in[10];
  const float* ln1b  = (const float*)d_in[11];
  const float* ffw1  = (const float*)d_in[12];
  const float* ffb1  = (const float*)d_in[13];
  const float* ffw2  = (const float*)d_in[14];
  const float* ffb2  = (const float*)d_in[15];
  const float* ln2g  = (const float*)d_in[16];
  const float* ln2b  = (const float*)d_in[17];

  size_t off = 0;
  auto take = [&](size_t bytes) -> void* {
    void* p = (char*)d_ws + off;
    off += (bytes + 255) & ~(size_t)255;
    return p;
  };
  float* h      = (float*) take((size_t)kRows * kD * 4);
  bf16*  xn     = (bf16*)  take((size_t)kRows * kD * 2);
  bf16*  qb     = (bf16*)  take((size_t)kRows * kD * 2);
  bf16*  kb     = (bf16*)  take((size_t)kRows * kD * 2);
  bf16*  vb     = (bf16*)  take((size_t)kRows * kD * 2);
  bf16*  ab     = (bf16*)  take((size_t)kRows * kD * 2);
  float* scores = (float*) take((size_t)kB * kH * kS * kS * 4);
  bf16*  pbuf   = (bf16*)  take((size_t)kB * kH * kS * kS * 2);
  bf16*  ffi    = (bf16*)  take((size_t)kRows * kDff * 2);
  bf16*  wqkv   = (bf16*)  take((size_t)3 * kD * kD * 2);
  bf16*  wout   = (bf16*)  take((size_t)kD * kD * 2);
  bf16*  wff1   = (bf16*)  take((size_t)kD * kDff * 2);
  bf16*  wff2   = (bf16*)  take((size_t)kDff * kD * 2);

  auto conv = [&](const float* src, bf16* dst, int n) {
    f32_to_bf16_k<<<(n + 255) / 256, 256, 0, stream>>>(src, dst, n);
  };
  auto gemm = [&](const bf16* A, int lda, long long sAb, long long sAh,
                  const bf16* B, int ldb, long long sBb, long long sBh, int btrans,
                  const float* bias, const float* resid, float* Cf, bf16* Cb,
                  int ldc, long long sCb, long long sCh,
                  int M, int N, int K, int Hdim, int nbatch, float alpha, int act) {
    dim3 grid((N + 127) / 128, (M + 127) / 128, nbatch);
    gemm_bf16_wmma<<<grid, 256, 0, stream>>>(A, lda, sAb, sAh, B, ldb, sBb, sBh, btrans,
                                             bias, resid, Cf, Cb, ldc, sCb, sCh,
                                             M, N, K, Hdim, alpha, act);
  };

  // Embedding
  embed_k<<<kRows, 256, 0, stream>>>(x, doy, w_in, b_in, pe, h);

  for (int i = 0; i < kL; ++i) {
    // stage this layer's weights as bf16
    conv(qkv_w + (size_t)i * 3 * kD * kD, wqkv, 3 * kD * kD);
    conv(out_w + (size_t)i * kD * kD,     wout, kD * kD);
    conv(ffw1  + (size_t)i * kD * kDff,   wff1, kD * kDff);
    conv(ffw2  + (size_t)i * kDff * kD,   wff2, kDff * kD);

    // xn = LN1(h)
    layernorm_k<<<kRows, 256, 0, stream>>>(h, ln1g + (size_t)i * kD, ln1b + (size_t)i * kD, xn);

    // Q, K, V projections
    for (int j = 0; j < 3; ++j) {
      bf16* outp = (j == 0) ? qb : (j == 1) ? kb : vb;
      gemm(xn, kD, 0, 0, wqkv + (size_t)j * kD * kD, kD, 0, 0, 0,
           qkv_b + ((size_t)i * 3 + j) * kD, nullptr, nullptr, outp,
           kD, 0, 0, kRows, kD, kD, 1, 1, 1.0f, 0);
    }

    // scores[b,h] = (Q . K^T) / sqrt(dk)   (batched over B*H, B transposed)
    gemm(qb, kD, (long long)kS * kD, kDk,
         kb, kD, (long long)kS * kD, kDk, 1,
         nullptr, nullptr, scores, nullptr,
         kS, (long long)kH * kSS, kSS,
         kS, kS, kDk, kH, kB * kH, 0.125f, 0);

    // softmax over the HEAD axis (faithful quirk) + mask
    softmax_head_k<<<(kB * kS * kS + 255) / 256, 256, 0, stream>>>(scores, mask, pbuf);

    // a[b,h] = P @ V  -> written into [B,S,D] at column h*64
    gemm(pbuf, kS, (long long)kH * kSS, kSS,
         vb, kD, (long long)kS * kD, kDk, 0,
         nullptr, nullptr, nullptr, ab,
         kD, (long long)kS * kD, kDk,
         kS, kDk, kS, kH, kB * kH, 1.0f, 0);

    // h = h + a @ out_w + out_b
    gemm(ab, kD, 0, 0, wout, kD, 0, 0, 0,
         out_b + (size_t)i * kD, h, h, nullptr,
         kD, 0, 0, kRows, kD, kD, 1, 1, 1.0f, 0);

    // xn = LN2(h)
    layernorm_k<<<kRows, 256, 0, stream>>>(h, ln2g + (size_t)i * kD, ln2b + (size_t)i * kD, xn);

    // ffi = gelu(xn @ ff_w1 + ff_b1)
    gemm(xn, kD, 0, 0, wff1, kDff, 0, 0, 0,
         ffb1 + (size_t)i * kDff, nullptr, nullptr, ffi,
         kDff, 0, 0, kRows, kDff, kD, 1, 1, 1.0f, 1);

    // h = h + ffi @ ff_w2 + ff_b2
    gemm(ffi, kDff, 0, 0, wff2, kD, 0, 0, 0,
         ffb2 + (size_t)i * kD, h, h, nullptr,
         kD, 0, 0, kRows, kD, kDff, 1, 1, 1.0f, 0);
  }

  copy_f32_k<<<(kRows * kD + 255) / 256, 256, 0, stream>>>(h, (float*)d_out, kRows * kD);
}
